// LeNet5_13786845020602
// MI455X (gfx1250) — compile-verified, matrix operand order checked by reading, not run on hardware
//
#include <hip/hip_runtime.h>
#include <hip/hip_bf16.h>
#include <math.h>

typedef __attribute__((ext_vector_type(16))) _Float16 v16h;
typedef __attribute__((ext_vector_type(8)))  _Float16 v8h;
typedef __attribute__((ext_vector_type(4)))  _Float16 v4h;
typedef __attribute__((ext_vector_type(2)))  _Float16 v2h;
typedef __attribute__((ext_vector_type(8)))  float    v8f;
typedef __attribute__((__vector_size__(4 * sizeof(int)))) int i32x4_;

#define WMMA(a, b, c) __builtin_amdgcn_wmma_f32_16x16x32_f16(false, (a), false, (b), (short)0, (c), false, false)

// --- gfx1250 async global->LDS copy (ASYNCcnt) with safe fallback -----------
#if defined(__has_builtin)
#if __has_builtin(__builtin_amdgcn_global_load_async_to_lds_b128) && \
    __has_builtin(__builtin_amdgcn_s_wait_asynccnt)
#define HAVE_ASYNC_LDS 1
#endif
#endif
#ifndef HAVE_ASYNC_LDS
#define HAVE_ASYNC_LDS 0
#endif

#if HAVE_ASYNC_LDS
#define ASYNC_B128(gsrc, ldst)                                          \
  __builtin_amdgcn_global_load_async_to_lds_b128(                       \
      (__attribute__((address_space(1))) i32x4_*)(gsrc),                \
      (__attribute__((address_space(3))) i32x4_*)(ldst), 0, 0)
#define ASYNC_WAIT() __builtin_amdgcn_s_wait_asynccnt(0)
#else
#define ASYNC_B128(gsrc, ldst) do { *(v8h*)(ldst) = *(const v8h*)(gsrc); } while (0)
#define ASYNC_WAIT() do { } while (0)
#endif

static __device__ const int IDX3c[6][3] = {{0,1,2},{1,2,3},{2,3,4},{3,4,5},{0,4,5},{0,1,5}};
static __device__ const int IDX4c[9][4] = {{0,1,2,3},{1,2,3,4},{2,3,4,5},{0,3,4,5},{0,1,4,5},{0,1,2,5},{0,1,3,4},{1,2,4,5},{0,2,3,5}};

// One 16-bit WMMA A/B operand fragment from a row-major tile: two aligned
// 128-bit loads at halves [h*8..+7] and [16+h*8..+7] (caller folds m*stride
// and h*8 into p). Row stride must be a multiple of 8 halves.
__device__ __forceinline__ v16h frag_ld(const _Float16* p) {
  v8h lo = *(const v8h*)p;
  v8h hi = *(const v8h*)(p + 16);
  return __builtin_shufflevector(lo, hi, 0, 1, 2, 3, 4, 5, 6, 7, 8, 9, 10, 11, 12, 13, 14, 15);
}

// ---------------------------------------------------------------------------
// Kernel 0: weight conversion/padding to f16 row-major GEMM layouts + zero stats
// ---------------------------------------------------------------------------
__global__ void lenet_prep(const float* __restrict__ conv1_w,
                           const float* __restrict__ c33w, const float* __restrict__ c34w,
                           const float* __restrict__ c36w,
                           const float* __restrict__ conv5_w,
                           const float* __restrict__ fc6_w, const float* __restrict__ fc7_w,
                           float* __restrict__ stats,
                           _Float16* __restrict__ W1h, _Float16* __restrict__ W3h,
                           _Float16* __restrict__ W5h, _Float16* __restrict__ W6h,
                           _Float16* __restrict__ W7h) {
  const int tid = threadIdx.x;
#pragma unroll 1
  for (int i = tid; i < 44; i += 256) stats[i] = 0.0f;
#pragma unroll 1
  for (int i = tid; i < 16 * 96; i += 256) {          // conv1: 16 x 96 (6 x 75 used)
    int oc = i / 96, k = i - oc * 96;
    float v = 0.0f;
    if (oc < 6 && k < 75) v = conv1_w[oc * 75 + k];
    W1h[i] = (_Float16)v;
  }
#pragma unroll 1
  for (int i = tid; i < 16 * 160; i += 256) {         // c3 expanded dense: 16 x 160 (150 used)
    int oc = i / 160, k = i - oc * 160;
    float v = 0.0f;
    if (k < 150) {
      int c = k / 25, r = k - c * 25;
      if (oc < 6) {
        for (int j = 0; j < 3; ++j) if (IDX3c[oc][j] == c) v = c33w[j * 25 + r];
      } else if (oc < 15) {
        for (int j = 0; j < 4; ++j) if (IDX4c[oc - 6][j] == c) v = c34w[j * 25 + r];
      } else {
        v = c36w[c * 25 + r];
      }
    }
    W3h[i] = (_Float16)v;
  }
#pragma unroll 1
  for (int i = tid; i < 128 * 416; i += 256) {        // conv5: 128 x 416 (120 x 400 used)
    int n = i / 416, k = i - n * 416;
    float v = 0.0f;
    if (n < 120 && k < 400) v = conv5_w[n * 400 + k];
    W5h[i] = (_Float16)v;
  }
#pragma unroll 1
  for (int i = tid; i < 96 * 128; i += 256) {         // fc6: 96 x 128 (84 x 120 used)
    int n = i / 128, k = i - n * 128;
    float v = 0.0f;
    if (n < 84 && k < 120) v = fc6_w[n * 120 + k];
    W6h[i] = (_Float16)v;
  }
#pragma unroll 1
  for (int i = tid; i < 16 * 96; i += 256) {          // fc7: 16 x 96 (10 x 84 used)
    int n = i / 96, k = i - n * 96;
    float v = 0.0f;
    if (n < 10 && k < 84) v = fc7_w[n * 84 + k];
    W7h[i] = (_Float16)v;
  }
}

// ---------------------------------------------------------------------------
// Kernel 1: conv1 (3->6, 5x5, 32x32 -> 28x28) via WMMA + BN stats.
// One block per image. Weights arrive via async-to-LDS DMA while the wave
// converts the f32 image to f16 with b128 loads. Then 7 chunks of 8 pixel-
// tiles: branchless im2col staging + 2x ds_load_b128 per WMMA fragment.
// ---------------------------------------------------------------------------
__global__ void lenet_conv1(const float* __restrict__ x, const _Float16* __restrict__ W1h,
                            const float* __restrict__ conv1_b,
                            _Float16* __restrict__ y1, float* __restrict__ stats) {
  __shared__ __align__(16) _Float16 simg[3 * 32 * 32];   // [c][y][x]
  __shared__ __align__(16) _Float16 swt[16 * 96];        // row-major N x K
  __shared__ __align__(16) _Float16 sA[8 * 16 * 96];     // 8 tiles, row-major 16 x 96
  __shared__ float ssum[6], ssq[6];
  const int b = blockIdx.x;
  const int tid = threadIdx.x;
  const float* xb = x + (size_t)b * 3072;
  __builtin_prefetch(xb + (tid << 3), 0, 1);             // global_prefetch_b8
  for (int i = tid; i < 192; i += 256) ASYNC_B128(W1h + i * 8, swt + i * 8);
  for (int i = tid; i < 768; i += 256) {                 // f32x4 load -> f16x4 LDS store
    const float4 q = ((const float4*)xb)[i];
    v4h h4 = {(_Float16)q.x, (_Float16)q.y, (_Float16)q.z, (_Float16)q.w};
    *(v4h*)&simg[i * 4] = h4;
  }
  if (tid < 6) { ssum[tid] = 0.0f; ssq[tid] = 0.0f; }
  ASYNC_WAIT();

  const int wave = tid >> 5, lane = tid & 31;
  const int nl = lane & 15, half = lane >> 4;
  const float bias = (nl < 6) ? conv1_b[nl] : 0.0f;
  float csum = 0.0f, csq = 0.0f;

  for (int chunk = 0; chunk < 7; ++chunk) {
    __syncthreads();
#pragma unroll 2
    for (int i = tid; i < 8 * 16 * 96; i += 256) {       // branchless im2col scatter
      const int tt = i / 1536, r0 = i - tt * 1536;
      const int m = r0 / 96, k = r0 - m * 96;
      const int p = chunk * 128 + tt * 16 + m;
      const int c = k / 25, rr = k - c * 25, kh = rr / 5, kw = rr - kh * 5;
      const int oy = p / 28, ox = p - oy * 28;
      const bool valid = (p < 784) & (k < 75);
      const int addr = valid ? (c * 1024 + (oy + kh) * 32 + ox + kw) : 0;
      const _Float16 v = simg[addr];                     // unconditional ds_load
      sA[i] = valid ? v : (_Float16)0.0f;                // v_cndmask, no saveexec
    }
    __syncthreads();
    const int t = chunk * 8 + wave;
    v8f acc = {};
#pragma unroll
    for (int kb = 0; kb < 3; ++kb) {
      const v16h a  = frag_ld(&sA[wave * 1536 + nl * 96 + kb * 32 + half * 8]);
      const v16h bb = frag_ld(&swt[nl * 96 + kb * 32 + half * 8]);
      acc = WMMA(a, bb, acc);
    }
    if (nl < 6) {
#pragma unroll
      for (int r = 0; r < 8; ++r) {
        const int p = t * 16 + r + half * 8;             // D row m = r + 8*half
        if (p < 784) {
          const float val = acc[r] + bias;
          y1[((size_t)b * 6 + nl) * 784 + p] = (_Float16)val;
          csum += val; csq += val * val;
        }
      }
    }
  }
  if (nl < 6) { atomicAdd(&ssum[nl], csum); atomicAdd(&ssq[nl], csq); }
  __syncthreads();
  if (tid < 6) { atomicAdd(&stats[tid], ssum[tid]); atomicAdd(&stats[6 + tid], ssq[tid]); }
}

// ---------------------------------------------------------------------------
// Kernel 2: BN finalize: scale = g*rsqrt(var+eps), shift = b - mean*scale
// ---------------------------------------------------------------------------
__global__ void lenet_bnfin(const float* __restrict__ sum, const float* __restrict__ sq,
                            const float* __restrict__ g, const float* __restrict__ bb,
                            float* __restrict__ scale, float* __restrict__ shift,
                            int nch, float inv_count) {
  const int c = threadIdx.x;
  if (c < nch) {
    const float mean = sum[c] * inv_count;
    const float var = sq[c] * inv_count - mean * mean;
    const float s = g[c] * rsqrtf(var + 1e-5f);
    scale[c] = s;
    shift[c] = bb[c] - mean * s;
  }
}

// ---------------------------------------------------------------------------
// Kernel 3: fused BN + 2x2 maxpool (scale>0 => affine-after-max equivalent)
// ---------------------------------------------------------------------------
__global__ void lenet_bnpool(const _Float16* __restrict__ in, _Float16* __restrict__ out,
                             const float* __restrict__ sc, const float* __restrict__ sh,
                             int C, int H, int W, long long total) {
  const long long i = (long long)blockIdx.x * 256 + threadIdx.x;
  if (i >= total) return;
  const int W2 = W >> 1, H2 = H >> 1;
  int xo = (int)(i % W2); long long t = i / W2;
  int yo = (int)(t % H2); t /= H2;
  int c = (int)(t % C);   long long b = t / C;
  const _Float16* p = in + (((b * C + c) * H + 2 * yo) * W + 2 * xo);
  const v2h r0 = *(const v2h*)p;                   // 4-byte aligned pair loads
  const v2h r1 = *(const v2h*)(p + W);
  const float m = fmaxf(fmaxf((float)r0[0], (float)r0[1]), fmaxf((float)r1[0], (float)r1[1]));
  out[i] = (_Float16)(m * sc[c] + sh[c]);
}

// ---------------------------------------------------------------------------
// Kernel 4: c3 dense 6->16, 5x5, 14x14 -> 10x10 via WMMA (K=160) + BN stats.
// Image tile and weights both arrive via async-to-LDS DMA.
// ---------------------------------------------------------------------------
__global__ void lenet_c3(const _Float16* __restrict__ h1, const _Float16* __restrict__ W3h,
                         const float* __restrict__ b3, const float* __restrict__ b4,
                         const float* __restrict__ b6,
                         _Float16* __restrict__ y3, float* __restrict__ stats) {
  __shared__ __align__(16) _Float16 simg[6 * 14 * 14];   // 1176 = 147 x 8 halves
  __shared__ __align__(16) _Float16 swt[16 * 160];
  __shared__ __align__(16) _Float16 sA[8 * 16 * 160];
  __shared__ float ssum[16], ssq[16];
  const int b = blockIdx.x;
  const int tid = threadIdx.x;
  const _Float16* hb = h1 + (size_t)b * 1176;
  for (int i = tid; i < 147; i += 256) ASYNC_B128(hb + i * 8, simg + i * 8);
  for (int i = tid; i < 320; i += 256) ASYNC_B128(W3h + i * 8, swt + i * 8);
  if (tid < 16) { ssum[tid] = 0.0f; ssq[tid] = 0.0f; }
  ASYNC_WAIT();
  __syncthreads();

#pragma unroll 2
  for (int i = tid; i < 8 * 16 * 160; i += 256) {        // branchless im2col scatter
    const int tt = i / 2560, r0 = i - tt * 2560;
    const int m = r0 / 160, k = r0 - m * 160;
    const int p = tt * 16 + m;
    const int c = k / 25, rr = k - c * 25, kh = rr / 5, kw = rr - kh * 5;
    const int oy = p / 10, ox = p - oy * 10;
    const bool valid = (p < 100) & (k < 150);
    const int addr = valid ? (c * 196 + (oy + kh) * 14 + ox + kw) : 0;
    const _Float16 v = simg[addr];
    sA[i] = valid ? v : (_Float16)0.0f;
  }
  __syncthreads();

  const int wave = tid >> 5, lane = tid & 31;
  const int nl = lane & 15, half = lane >> 4;
  const float bias = (nl < 6) ? b3[0] : ((nl < 15) ? b4[0] : b6[0]);
  float csum = 0.0f, csq = 0.0f;

  v8f acc = {};
#pragma unroll
  for (int kb = 0; kb < 5; ++kb) {
    const v16h a  = frag_ld(&sA[wave * 2560 + nl * 160 + kb * 32 + half * 8]);
    const v16h bb = frag_ld(&swt[nl * 160 + kb * 32 + half * 8]);
    acc = WMMA(a, bb, acc);
  }
#pragma unroll
  for (int r = 0; r < 8; ++r) {
    const int p = wave * 16 + r + half * 8;
    if (p < 100) {
      const float val = acc[r] + bias;
      y3[((size_t)b * 16 + nl) * 100 + p] = (_Float16)val;
      csum += val; csq += val * val;
    }
  }
  atomicAdd(&ssum[nl], csum);
  atomicAdd(&ssq[nl], csq);
  __syncthreads();
  if (tid < 16) { atomicAdd(&stats[12 + tid], ssum[tid]); atomicAdd(&stats[28 + tid], ssq[tid]); }
}

// ---------------------------------------------------------------------------
// Kernel 5: fused head. Per block: 16 samples.
// conv5 GEMM 16x416x128 -> fc6 16x128x96 -> fc7 16x96x16 -> sigmoid.
// Feature rows DMA'd to LDS async; every fragment is two 128-bit loads.
// ---------------------------------------------------------------------------
__global__ void lenet_head(const _Float16* __restrict__ f, const _Float16* __restrict__ W5h,
                           const _Float16* __restrict__ W6h, const _Float16* __restrict__ W7h,
                           const float* __restrict__ b5, const float* __restrict__ b6,
                           const float* __restrict__ b7, float* __restrict__ out) {
  __shared__ __align__(16) _Float16 sf[16 * 416];
  __shared__ __align__(16) _Float16 a1[16 * 128];
  __shared__ __align__(16) _Float16 a2[16 * 96];
  const int tid = threadIdx.x;
  const size_t base = (size_t)blockIdx.x * 16;
  for (int i = tid; i < 16 * 50; i += 256) {             // 50 b128 chunks per sample row
    const int s = i / 50, j = i - s * 50;
    ASYNC_B128(f + (base + s) * 400 + j * 8, sf + s * 416 + j * 8);
  }
  {                                                      // zero the 400..415 K-pad
    const int s = tid >> 4, k = 400 + (tid & 15);
    sf[s * 416 + k] = (_Float16)0.0f;
  }
  ASYNC_WAIT();
  __syncthreads();
  const int wave = tid >> 5, lane = tid & 31;
  const int nl = lane & 15, half = lane >> 4;

  { // conv5: 8 waves, one 16-col N-tile each
    const int n = wave * 16 + nl;
    v8f acc = {};
#pragma unroll
    for (int kb = 0; kb < 13; ++kb) {
      const v16h a  = frag_ld(&sf[nl * 416 + kb * 32 + half * 8]);
      const v16h bb = frag_ld(&W5h[n * 416 + kb * 32 + half * 8]);
      acc = WMMA(a, bb, acc);
    }
    const float bias = (n < 120) ? b5[n] : 0.0f;
#pragma unroll
    for (int r = 0; r < 8; ++r) a1[(r + half * 8) * 128 + n] = (_Float16)(acc[r] + bias);
  }
  __syncthreads();
  if (wave < 6) { // fc6
    const int n = wave * 16 + nl;
    v8f acc = {};
#pragma unroll
    for (int kb = 0; kb < 4; ++kb) {
      const v16h a  = frag_ld(&a1[nl * 128 + kb * 32 + half * 8]);
      const v16h bb = frag_ld(&W6h[n * 128 + kb * 32 + half * 8]);
      acc = WMMA(a, bb, acc);
    }
    const float bias = (n < 84) ? b6[n] : 0.0f;
#pragma unroll
    for (int r = 0; r < 8; ++r) a2[(r + half * 8) * 96 + n] = (_Float16)(acc[r] + bias);
  }
  __syncthreads();
  if (wave == 0) { // fc7 + sigmoid
    v8f acc = {};
#pragma unroll
    for (int kb = 0; kb < 3; ++kb) {
      const v16h a  = frag_ld(&a2[nl * 96 + kb * 32 + half * 8]);
      const v16h bb = frag_ld(&W7h[nl * 96 + kb * 32 + half * 8]);
      acc = WMMA(a, bb, acc);
    }
    if (nl < 10) {
      const float bias = b7[nl];
#pragma unroll
      for (int r = 0; r < 8; ++r) {
        const float val = acc[r] + bias;
        out[(base + r + half * 8) * 10 + nl] = 1.0f / (1.0f + expf(-val));
      }
    }
  }
}

// ---------------------------------------------------------------------------
// Host launcher
// ---------------------------------------------------------------------------
extern "C" void kernel_launch(void* const* d_in, const int* in_sizes, int n_in,
                              void* d_out, int out_size, void* d_ws, size_t ws_size,
                              hipStream_t stream) {
  const float* x       = (const float*)d_in[0];
  const float* conv1_w = (const float*)d_in[1];
  const float* conv1_b = (const float*)d_in[2];
  const float* bn1_g   = (const float*)d_in[3];
  const float* bn1_b   = (const float*)d_in[4];
  const float* c33w    = (const float*)d_in[5];
  const float* c33b    = (const float*)d_in[6];
  const float* c34w    = (const float*)d_in[7];
  const float* c34b    = (const float*)d_in[8];
  const float* c36w    = (const float*)d_in[9];
  const float* c36b    = (const float*)d_in[10];
  const float* bn3_g   = (const float*)d_in[11];
  const float* bn3_b   = (const float*)d_in[12];
  const float* conv5_w = (const float*)d_in[13];
  const float* conv5_b = (const float*)d_in[14];
  const float* fc6_w   = (const float*)d_in[15];
  const float* fc6_b   = (const float*)d_in[16];
  const float* fc7_w   = (const float*)d_in[17];
  const float* fc7_b   = (const float*)d_in[18];
  float* out = (float*)d_out;

  char* ws = (char*)d_ws;
  float*    stats = (float*)(ws + 0);            // 44 floats: sum1[6] sq1[6] sum3[16] sq3[16]
  float*    ss1   = (float*)(ws + 256);          // scale1[6], shift1[6]
  float*    ss3   = (float*)(ws + 512);          // scale3[16], shift3[16]
  _Float16* W1h   = (_Float16*)(ws + 768);
  _Float16* W3h   = (_Float16*)(ws + 3840);
  _Float16* W5h   = (_Float16*)(ws + 8960);
  _Float16* W6h   = (_Float16*)(ws + 115456);
  _Float16* W7h   = (_Float16*)(ws + 140032);
  // Region R0 (77.1 MB) holds y1, then is reused for y3 (26.2 MB) + features (6.6 MB).
  _Float16* Y1    = (_Float16*)(ws + 143360);
  _Float16* Y3    = Y1;                                     // y1 dead by then
  _Float16* F     = (_Float16*)(ws + 143360 + 26214400ULL);
  _Float16* H1    = (_Float16*)(ws + 143360 + 77070336ULL); // R1 (19.3 MB)

  const int B = 8192;

  lenet_prep<<<1, 256, 0, stream>>>(conv1_w, c33w, c34w, c36w, conv5_w, fc6_w, fc7_w,
                                    stats, W1h, W3h, W5h, W6h, W7h);

  lenet_conv1<<<B, 256, 0, stream>>>(x, W1h, conv1_b, Y1, stats);

  lenet_bnfin<<<1, 32, 0, stream>>>(stats + 0, stats + 6, bn1_g, bn1_b,
                                    ss1, ss1 + 6, 6, 1.0f / (float)(B * 784));

  {
    const long long total = (long long)B * 6 * 14 * 14;
    lenet_bnpool<<<(unsigned)((total + 255) / 256), 256, 0, stream>>>(
        Y1, H1, ss1, ss1 + 6, 6, 28, 28, total);
  }

  lenet_c3<<<B, 256, 0, stream>>>(H1, W3h, c33b, c34b, c36b, Y3, stats);

  lenet_bnfin<<<1, 32, 0, stream>>>(stats + 12, stats + 28, bn3_g, bn3_b,
                                    ss3, ss3 + 16, 16, 1.0f / (float)(B * 100));

  {
    const long long total = (long long)B * 16 * 5 * 5;
    lenet_bnpool<<<(unsigned)((total + 255) / 256), 256, 0, stream>>>(
        Y3, F, ss3, ss3 + 16, 16, 10, 10, total);
  }

  lenet_head<<<B / 16, 256, 0, stream>>>(F, W5h, W6h, W7h, conv5_b, fc6_b, fc7_b, out);
}
